// LSTM_13563506721309
// MI455X (gfx1250) — matrix operand order repaired and checked by reading, not compiled, not run on hardware
//
#include <hip/hip_runtime.h>
#include <stdint.h>

#define BATCH 128
#define SEQ   256
#define HID   1024
#define DCOL  1025
#define NCLS  10
#define NWG   32                  // persistent workgroups
#define COLS_PER_WG (HID / NWG)   // 32 hidden units per WG
#define NTILES_TOTAL (HID / 16)   // 64 N-tiles per gate
#define KBLKS (HID / 32)          // 32 K-blocks of 32
#define FRAG_BYTES   1024         // one 32x16 bf16 fragment (512 elems)
#define REGION_BYTES (KBLKS * FRAG_BYTES)       // 32 KB per (gate, n-subtile)
#define LDS_BYTES    (8 * REGION_BYTES)         // 256 KB per WG

typedef __attribute__((ext_vector_type(16))) __bf16 v16bf;
typedef __attribute__((ext_vector_type(8)))  float  v8f;

// ---------- helpers ----------
__device__ __forceinline__ uint16_t f2bf(float f) {
    unsigned u = __float_as_uint(f);
    u += 0x7FFFu + ((u >> 16) & 1u);   // round-to-nearest-even
    return (uint16_t)(u >> 16);
}
__device__ __forceinline__ float bf2f(uint16_t h) {
    return __uint_as_float(((unsigned)h) << 16);
}
__device__ __forceinline__ float sigmoid_f(float v) {
    return 1.0f / (1.0f + __expf(-v));
}
__device__ __forceinline__ float tanh_f(float v) {
    return 2.0f / (1.0f + __expf(-2.0f * v)) - 1.0f;   // saturates at +-1
}

// ---------- prologue: zero h0 buffer + barrier counter ----------
__global__ void LSTM_init_state(uint32_t* __restrict__ h0w, unsigned* __restrict__ cnt) {
    unsigned tid = blockIdx.x * blockDim.x + threadIdx.x;
    if (tid < (BATCH * HID / 2)) h0w[tid] = 0u;      // h0 = 0 (bf16 pairs)
    if (tid == 0) *cnt = 0u;                          // reset grid barrier
}

// ---------- prologue: pack 4 gate weight matrices into WMMA B-fragment layout ----------
// Per (gate, ntile, kblock): 512 bf16 stored as 2 chunks of [lane(32)][8 elems].
// Fragment semantics: B[k][n], n = ntile*16 + lane%16,
//                     k = kblock*32 + (lane>=16 ? 16 : 0) + e,  e = chunk*8 + e'.
// Sourced from W_gate[n][1 + k]  (column 0 = x weight, handled in epilogue).
__global__ void LSTM_pack_weights(const float* __restrict__ wg, const float* __restrict__ wi,
                                  const float* __restrict__ wf, const float* __restrict__ wo,
                                  uint16_t* __restrict__ packed) {
    unsigned tid  = blockIdx.x * blockDim.x + threadIdx.x;   // 0 .. 262143
    unsigned lane = tid & 31u;
    unsigned j    = (((tid >> 10) & 63u) * 16u) + (lane & 15u);
    unsigned kbase = (((tid >> 5) & 31u) * 32u) + ((lane & 16u) ? 16u : 0u);
    unsigned g    = tid >> 16;
    const float* W = (g == 0) ? wg : (g == 1) ? wi : (g == 2) ? wf : wo;
    const float* src = W + (size_t)j * DCOL + 1 + kbase;
    uint16_t* dst = packed + (size_t)(tid >> 5) * 512;       // fragment base
#pragma unroll
    for (int e = 0; e < 16; ++e)
        dst[((e >> 3) << 8) + lane * 8 + (e & 7)] = f2bf(src[e]);
}

// ---------- persistent LSTM kernel ----------
__global__ void __launch_bounds__(256, 1)
LSTM_persistent(const float* __restrict__ x,
                const uint16_t* __restrict__ packedW,
                const float* __restrict__ wg, const float* __restrict__ bg,
                const float* __restrict__ wi, const float* __restrict__ bi,
                const float* __restrict__ wf, const float* __restrict__ bf_,
                const float* __restrict__ wo, const float* __restrict__ bo,
                uint16_t* __restrict__ h0, uint16_t* __restrict__ h1,
                unsigned* __restrict__ cnt) {
    extern __shared__ uint8_t lds_raw[];                 // 256 KB: 8 regions x 32 KB

    const int tid    = threadIdx.x;
    const int lane   = tid & 31;
    const int wave   = tid >> 5;            // 0..7 -> batch rows [16w,16w+16)
    const int lanelo = lane & 15;
    const int lanehi = (lane >> 4) & 1;
    const int m0     = wave * 16;
    const int colbase = blockIdx.x * COLS_PER_WG;

    // ---- one-time: stage this WG's 256 KB weight slice into LDS (step-invariant) ----
#pragma unroll
    for (int f = 0; f < 8; ++f) {
        const uint4* src = (const uint4*)(packedW
            + (((size_t)(f >> 1) * NTILES_TOTAL + (size_t)blockIdx.x * 2 + (f & 1)) * KBLKS) * 512);
        uint4* dst = (uint4*)(lds_raw + f * REGION_BYTES);
        for (int i = tid; i < REGION_BYTES / 16; i += 256)   // 2048 uint4 per region
            dst[i] = src[i];
    }
    __syncthreads();

    // Hoist per-lane step-invariant scalars: biases + x-column weights.
    float bgv[2], biv[2], bfv[2], bov[2];
    float w0g[2], w0i[2], w0f[2], w0o[2];
#pragma unroll
    for (int nt = 0; nt < 2; ++nt) {
        int col = colbase + nt * 16 + lanelo;
        bgv[nt] = bg[col];  biv[nt] = bi[col];  bfv[nt] = bf_[col];  bov[nt] = bo[col];
        w0g[nt] = wg[(size_t)col * DCOL];
        w0i[nt] = wi[(size_t)col * DCOL];
        w0f[nt] = wf[(size_t)col * DCOL];
        w0o[nt] = wo[(size_t)col * DCOL];
    }

    // LDS byte offsets per fragment stream (f = gate*2 + nt).
    unsigned ldsoff[8];
#pragma unroll
    for (int f = 0; f < 8; ++f) ldsoff[f] = f * REGION_BYTES + lane * 16;

    const v8f vzero = {0.f, 0.f, 0.f, 0.f, 0.f, 0.f, 0.f, 0.f};
    v8f cst[2] = {vzero, vzero};            // cell state lives in registers all 256 steps

    union BU { uint4 q[2]; v16bf v; };

    for (int t = 0; t < SEQ; ++t) {
        const uint16_t* hprev = (t & 1) ? h1 : h0;
        uint16_t*       hnext = (t & 1) ? h0 : h1;

        v8f acc[8] = {vzero, vzero, vzero, vzero, vzero, vzero, vzero, vzero};

        // A-fragment row pointer for this lane (contiguous k = 16*half + s, matching packed B).
        const uint16_t* arow = hprev + (size_t)(m0 + lanelo) * HID + (lanehi ? 16 : 0);

        // Software-pipelined GEMM: ping-pong B fragments so ds_load latency is
        // covered by a full 8-WMMA group.
        BU Bp[8], Bq[8];
#pragma unroll
        for (int f = 0; f < 8; ++f) {
            unsigned o = ldsoff[f];
            Bp[f].q[0] = *(const uint4*)(lds_raw + o);
            Bp[f].q[1] = *(const uint4*)(lds_raw + o + 512);
        }

#pragma unroll
        for (int kb = 0; kb < KBLKS; kb += 2) {
            BU A0;
            A0.q[0] = *(const uint4*)(arow + kb * 32);
            A0.q[1] = *(const uint4*)(arow + kb * 32 + 8);
            if (kb + 2 < KBLKS)
                __builtin_prefetch(arow + (kb + 2) * 32, 0, 3);  // near-scope prefetch

#pragma unroll
            for (int f = 0; f < 8; ++f) {                         // B for kb+1
                unsigned o = ldsoff[f] + (unsigned)(kb + 1) * FRAG_BYTES;
                Bq[f].q[0] = *(const uint4*)(lds_raw + o);
                Bq[f].q[1] = *(const uint4*)(lds_raw + o + 512);
            }
#pragma unroll
            for (int f = 0; f < 8; ++f)
                acc[f] = __builtin_amdgcn_wmma_f32_16x16x32_bf16(
                             false, A0.v, false, Bp[f].v, (short)0, acc[f], false, false);

            BU A1;
            A1.q[0] = *(const uint4*)(arow + kb * 32 + 32);
            A1.q[1] = *(const uint4*)(arow + kb * 32 + 40);
            if (kb + 2 < KBLKS) {
#pragma unroll
                for (int f = 0; f < 8; ++f) {                     // B for kb+2
                    unsigned o = ldsoff[f] + (unsigned)(kb + 2) * FRAG_BYTES;
                    Bp[f].q[0] = *(const uint4*)(lds_raw + o);
                    Bp[f].q[1] = *(const uint4*)(lds_raw + o + 512);
                }
            }
#pragma unroll
            for (int f = 0; f < 8; ++f)
                acc[f] = __builtin_amdgcn_wmma_f32_16x16x32_bf16(
                             false, A1.v, false, Bq[f].v, (short)0, acc[f], false, false);
        }

        // Epilogue: gates -> activations -> cell update -> write bf16 h_next.
#pragma unroll
        for (int nt = 0; nt < 2; ++nt) {
#pragma unroll
            for (int r = 0; r < 8; ++r) {
                int row = m0 + r + lanehi * 8;
                int col = colbase + nt * 16 + lanelo;
                float xv = x[(size_t)row * SEQ + t];
                float gv = acc[0 + nt][r] + bgv[nt] + xv * w0g[nt];
                float iv = acc[2 + nt][r] + biv[nt] + xv * w0i[nt];
                float fv = acc[4 + nt][r] + bfv[nt] + xv * w0f[nt];
                float ov = acc[6 + nt][r] + bov[nt] + xv * w0o[nt];
                float cn = tanh_f(gv) * sigmoid_f(iv) + cst[nt][r] * sigmoid_f(fv);
                cst[nt][r] = cn;
                float hn = tanh_f(cn) * sigmoid_f(ov);
                hnext[(size_t)row * HID + col] = f2bf(hn);
            }
        }

        // Grid-wide barrier (monotonic counter; all 32 WGs are resident).
        __threadfence();
        __syncthreads();
        if (tid == 0) {
            __hip_atomic_fetch_add(cnt, 1u, __ATOMIC_ACQ_REL, __HIP_MEMORY_SCOPE_AGENT);
            unsigned target = (unsigned)(t + 1) * NWG;
            while (__hip_atomic_load(cnt, __ATOMIC_ACQUIRE, __HIP_MEMORY_SCOPE_AGENT) < target)
                __builtin_amdgcn_s_sleep(2);
        }
        __syncthreads();
    }
}

// ---------- final projection: out[b][c] = h_last[b] . wp[c] + bp[c] ----------
__global__ void LSTM_proj(const uint16_t* __restrict__ h, const float* __restrict__ wp,
                          const float* __restrict__ bp, float* __restrict__ out) {
    int idx = blockIdx.x * blockDim.x + threadIdx.x;
    if (idx >= BATCH * NCLS) return;
    int b = idx / NCLS, c = idx % NCLS;
    const uint16_t* hr = h + (size_t)b * HID;
    const float*    wr = wp + (size_t)c * HID;
    float s = 0.f;
    for (int j = 0; j < HID; ++j) s += bf2f(hr[j]) * wr[j];
    out[idx] = s + bp[c];
}

extern "C" void kernel_launch(void* const* d_in, const int* in_sizes, int n_in,
                              void* d_out, int out_size, void* d_ws, size_t ws_size,
                              hipStream_t stream) {
    (void)in_sizes; (void)n_in; (void)out_size; (void)ws_size;
    const float* x  = (const float*)d_in[0];
    const float* wg = (const float*)d_in[1];
    const float* bg = (const float*)d_in[2];
    const float* wi = (const float*)d_in[3];
    const float* bi = (const float*)d_in[4];
    const float* wf = (const float*)d_in[5];
    const float* bf = (const float*)d_in[6];
    const float* wo = (const float*)d_in[7];
    const float* bo = (const float*)d_in[8];
    const float* wp = (const float*)d_in[9];
    const float* bp = (const float*)d_in[10];
    float* out = (float*)d_out;

    char* ws = (char*)d_ws;
    unsigned* cnt     = (unsigned*)ws;                                   // 256 B slot
    uint16_t* h0      = (uint16_t*)(ws + 256);                           // 256 KB
    uint16_t* h1      = (uint16_t*)(ws + 256 + BATCH * HID * 2);         // 256 KB
    uint16_t* packedW = (uint16_t*)(ws + 256 + 2 * BATCH * HID * 2);     // 8 MB

    LSTM_init_state<<<256, 256, 0, stream>>>((uint32_t*)h0, cnt);
    LSTM_pack_weights<<<1024, 256, 0, stream>>>(wg, wi, wf, wo, packedW);
    LSTM_persistent<<<NWG, 256, LDS_BYTES, stream>>>(x, packedW, wg, bg, wi, bi, wf, bf,
                                                     wo, bo, h0, h1, cnt);
    LSTM_proj<<<(BATCH * NCLS + 255) / 256, 256, 0, stream>>>(h0, wp, bp, out);
}
